// RotaryMultiheadAttention_9079560864383
// MI455X (gfx1250) — matrix operand order repaired and checked by reading, not compiled, not run on hardware
//
#include <hip/hip_runtime.h>
#include <hip/hip_bf16.h>

typedef __attribute__((ext_vector_type(16))) _Float16 v16h;
typedef __attribute__((ext_vector_type(8)))  float    v8f;

#define WMMA_F32_F16(a, b, c) \
  __builtin_amdgcn_wmma_f32_16x16x32_f16(false, (a), false, (b), (short)0, (c), false, false)

// Problem constants: B=4, S=2048, E=1024, H=16, hd=64

// ---------------------------------------------------------------------------
// Kernel 0: f32 -> f16 conversion (weights + activations), one element per thread.
// ---------------------------------------------------------------------------
__global__ void cvt_f16_kernel(const float* __restrict__ src,
                               _Float16* __restrict__ dst, int n) {
  int idx = blockIdx.x * 256 + threadIdx.x;
  if (idx < n) dst[idx] = (_Float16)src[idx];
}

// ---------------------------------------------------------------------------
// Kernel 1: QKV projection, all-f16 operands.
// out = X @ W[zE:(z+1)E].T + bias, scattered into (B,H,S,hd) f16 layout.
// grid = (8192/128, 1024/64, 3), block = 256 (8 waves).
// Weight chunk (64 rows x 64 K, 8KB f16) staged in LDS, shared by all waves.
// ---------------------------------------------------------------------------
__global__ void qkv_proj_kernel(const _Float16* __restrict__ qin,
                                const _Float16* __restrict__ kin,
                                const _Float16* __restrict__ vin,
                                const _Float16* __restrict__ Wh,  // (3E, E) f16
                                const float* __restrict__ bin,    // (3E)
                                _Float16* __restrict__ qo,
                                _Float16* __restrict__ ko,
                                _Float16* __restrict__ vo) {
  __shared__ __align__(16) _Float16 Ws[64 * 64];

  const int tid  = threadIdx.x;
  const int lane = tid & 31;
  const int wave = tid >> 5;
  const int z = blockIdx.z;
  const _Float16* X = (z == 0) ? qin : (z == 1) ? kin : vin;
  _Float16* O       = (z == 0) ? qo  : (z == 1) ? ko  : vo;

  const int m0 = blockIdx.x * 128 + wave * 16;
  const int n0 = blockIdx.y * 64;

  v8f acc[4] = {};

  const int arow = lane & 15;
  const int akb  = (lane & 16) ? 16 : 0;  // A: lanes 16-31 hold K+16
  const int bn   = lane & 15;
  const int bkb  = (lane & 16) ? 16 : 0;  // B: lanes 16-31 hold K+16

  for (int kc = 0; kc < 1024; kc += 64) {
    __syncthreads();
    // stage W chunk: rows n0..n0+63, K = kc..kc+63 (4096 halves, 16B chunks)
    for (int i = tid; i < 512; i += 256) {
      int r = i >> 3;
      int cc = (i & 7) * 8;
      *(uint4*)&Ws[r * 64 + cc] =
          *(const uint4*)(Wh + (size_t)(z * 1024 + n0 + r) * 1024 + kc + cc);
    }
    __syncthreads();
#pragma unroll
    for (int kk = 0; kk < 64; kk += 32) {
      const _Float16* ap = X + (size_t)(m0 + arow) * 1024 + kc + kk + akb;
      v16h a;
#pragma unroll
      for (int i = 0; i < 8; ++i) { a[i] = ap[i]; a[8 + i] = ap[8 + i]; }
      // load all 4 B fragments first (independent regs -> pipelined ds loads)
      v16h bfr[4];
#pragma unroll
      for (int nt = 0; nt < 4; ++nt) {
        const _Float16* bp = Ws + (nt * 16 + bn) * 64 + kk + bkb;
#pragma unroll
        for (int i = 0; i < 16; ++i) bfr[nt][i] = bp[i];
      }
#pragma unroll
      for (int nt = 0; nt < 4; ++nt) acc[nt] = WMMA_F32_F16(a, bfr[nt], acc[nt]);
    }
  }

  const int crb = (lane & 16) ? 8 : 0;
#pragma unroll
  for (int nt = 0; nt < 4; ++nt) {
    int col = n0 + nt * 16 + (lane & 15);
    float bias = bin[z * 1024 + col];
    int h = col >> 6, d = col & 63;
#pragma unroll
    for (int r = 0; r < 8; ++r) {
      int row = m0 + crb + r;            // row in [0, B*S)
      int bb = row >> 11, s = row & 2047;
      O[((size_t)(bb * 16 + h) * 2048 + s) * 64 + d] = (_Float16)(acc[nt][r] + bias);
    }
  }
}

// ---------------------------------------------------------------------------
// Kernel 2: RoPE in-place on f16 q/k (q gets 1/8 scale folded in) + V
// transpose into d-major Vt.  One thread per (b,h,s,i) rotary pair.
// ---------------------------------------------------------------------------
__global__ void rope_pack_kernel(_Float16* __restrict__ Qr,
                                 _Float16* __restrict__ Kr,
                                 const _Float16* __restrict__ Vr,
                                 _Float16* __restrict__ Vt) {
  int idx = blockIdx.x * 256 + threadIdx.x;    // B*H*S*32 = 4194304 threads
  int i  = idx & 31;
  int s  = (idx >> 5) & 2047;
  int bh = idx >> 16;
  size_t base = ((size_t)(bh * 2048 + s)) * 64 + 2 * i;

  // inv_freq = 10000^(-2i/64) = exp(-i * ln(1e4)/32)
  float ang = (float)s * expf(-(float)i * 0.28782313662425572f);
  float c = cosf(ang), sn = sinf(ang);

  float qe = (float)Qr[base], qo2 = (float)Qr[base + 1];
  Qr[base]     = (_Float16)((qe * c - qo2 * sn) * 0.125f);
  Qr[base + 1] = (_Float16)((qo2 * c + qe * sn) * 0.125f);

  float ke = (float)Kr[base], ko2 = (float)Kr[base + 1];
  Kr[base]     = (_Float16)(ke * c - ko2 * sn);
  Kr[base + 1] = (_Float16)(ko2 * c + ke * sn);

  size_t vt0 = ((size_t)(bh * 64 + 2 * i)) * 2048 + s;
  Vt[vt0]        = Vr[base];
  Vt[vt0 + 2048] = Vr[base + 1];
}

// ---------------------------------------------------------------------------
// Kernel 3: fused attention per (b, 16-row q tile), looping all 16 heads.
// LDS: 16x2048 f32 head-mean accumulator (128KB) + 16x2048 f16 scores (64KB)
//      + 16x2048 f16 probs (64KB) + reductions + 16x64 AV stage
//      = 267392 bytes (CDNA5 WGP has 320KB).
// grid = B*(S/16) = 512 blocks, block = 256 (8 waves).
// ---------------------------------------------------------------------------
__global__ void attention_kernel(const _Float16* __restrict__ Qh,
                                 const _Float16* __restrict__ Kh,
                                 const _Float16* __restrict__ Vt,
                                 _Float16* __restrict__ raw,   // (B*S, E) f16
                                 float* __restrict__ wout) {   // (B, Sq, Sk) f32
  extern __shared__ float smem[];
  float*     wa    = smem;                            // 32768 f32
  _Float16*  sch   = (_Float16*)(smem + 32768);       // 32768 f16 (scores)
  _Float16*  ph    = (_Float16*)(smem + 49152);       // 32768 f16 (probs)
  float*     part  = smem + 65536;                    // 256
  float*     rmax  = smem + 65792;                    // 16
  float*     rinv  = smem + 65808;                    // 16
  float*     outst = smem + 65824;                    // 1024

  const int tid  = threadIdx.x;
  const int lane = tid & 31;
  const int wave = tid >> 5;
  const int b  = blockIdx.x >> 7;
  const int q0 = (blockIdx.x & 127) * 16;

  for (int i = tid; i < 32768; i += 256) wa[i] = 0.0f;
  __syncthreads();

  const int arow = lane & 15;
  const int akb  = (lane & 16) ? 16 : 0;
  const int bn   = lane & 15;
  const int bkb  = (lane & 16) ? 16 : 0;
  const int ccol = lane & 15;
  const int crb  = (lane & 16) ? 8 : 0;
  const int srow = tid >> 4;   // softmax: 16 rows x 16 segments of 128 cols
  const int sseg = tid & 15;

  for (int h = 0; h < 16; ++h) {
    const _Float16* Qp = Qh + ((size_t)((b * 16 + h) * 2048 + q0)) * 64;
    const _Float16* Kp = Kh + ((size_t)((b * 16 + h) * 2048)) * 64;
    const _Float16* Vp = Vt + ((size_t)(b * 16 + h)) * 64 * 2048;

    // Q A-fragments for K=0..31 and K=32..63
    v16h aq0, aq1;
    {
      const _Float16* p0 = Qp + arow * 64 + akb;
#pragma unroll
      for (int i = 0; i < 8; ++i) {
        aq0[i] = p0[i];       aq0[8 + i] = p0[8 + i];
        aq1[i] = p0[32 + i];  aq1[8 + i] = p0[40 + i];
      }
    }

    // ---- scores: S = (q/8) @ K^T, 16 k-tiles per wave ----
#pragma unroll 2
    for (int kt = wave; kt < 128; kt += 8) {
      const _Float16* kp = Kp + (size_t)(kt * 16 + bn) * 64 + bkb;
      v16h b0, b1;
#pragma unroll
      for (int i = 0; i < 16; ++i) { b0[i] = kp[i]; b1[i] = kp[32 + i]; }
      v8f c = {};
      c = WMMA_F32_F16(aq0, b0, c);
      c = WMMA_F32_F16(aq1, b1, c);
      int col = kt * 16 + ccol;
#pragma unroll
      for (int r = 0; r < 8; ++r) sch[(crb + r) * 2048 + col] = (_Float16)c[r];
    }
    __syncthreads();

    // ---- softmax over 2048 cols per row ----
    const _Float16* srp = sch + srow * 2048 + sseg * 128;
    _Float16* phrow = ph + srow * 2048 + sseg * 128;
    float mx = -3.0e38f;
    for (int j = 0; j < 128; ++j) mx = fmaxf(mx, (float)srp[j]);
    part[tid] = mx;
    __syncthreads();
    if (sseg == 0) {
      float m2 = part[srow * 16];
      for (int t = 1; t < 16; ++t) m2 = fmaxf(m2, part[srow * 16 + t]);
      rmax[srow] = m2;
    }
    __syncthreads();
    float m2 = rmax[srow];
    float sum = 0.0f;
    for (int j = 0; j < 128; ++j) {
      float p = __expf((float)srp[j] - m2);
      phrow[j] = (_Float16)p;
      sum += p;
    }
    part[tid] = sum;
    __syncthreads();
    if (sseg == 0) {
      float s2 = 0.0f;
      for (int t = 0; t < 16; ++t) s2 += part[srow * 16 + t];
      rinv[srow] = 1.0f / s2;
    }
    for (int i = tid; i < 1024; i += 256) outst[i] = 0.0f;
    __syncthreads();

    // ---- accumulate head-mean of normalized weights ----
    {
      float f = rinv[srow] * 0.0625f;   // 1/sum * 1/H
      float* wrp = wa + srow * 2048 + sseg * 128;
      for (int j = 0; j < 128; ++j) wrp[j] += (float)phrow[j] * f;
    }

    // ---- AV: O = P(16x2048) @ V(2048x64); waves split 4 n-tiles x 2 K-halves
    const int nt = wave & 3, khalf = wave >> 2;
    v8f cav = {};
#pragma unroll 2
    for (int k0 = khalf * 1024; k0 < khalf * 1024 + 1024; k0 += 32) {
      const _Float16* pp = ph + arow * 2048 + k0 + akb;
      v16h ap;
#pragma unroll
      for (int i = 0; i < 8; ++i) { ap[i] = pp[i]; ap[8 + i] = pp[8 + i]; }
      const _Float16* vp = Vp + (size_t)(nt * 16 + bn) * 2048 + k0 + bkb;
      v16h bv;
#pragma unroll
      for (int i = 0; i < 16; ++i) bv[i] = vp[i];
      cav = WMMA_F32_F16(ap, bv, cav);
    }
    {
      int col = nt * 16 + ccol;
#pragma unroll
      for (int r = 0; r < 8; ++r) atomicAdd(&outst[(crb + r) * 64 + col], cav[r]);
    }
    __syncthreads();

    // ---- scale by 1/rowsum, write raw (B*S, E) f16 ----
    for (int i = tid; i < 1024; i += 256) {
      int r = i >> 6, d = i & 63;
      raw[((size_t)(b * 2048 + q0 + r)) * 1024 + h * 64 + d] =
          (_Float16)(outst[i] * rinv[r]);
    }
    __syncthreads();
  }

  // ---- write head-averaged attention weights ----
  for (int i = tid; i < 32768; i += 256) {
    int r = i >> 11, cidx = i & 2047;
    wout[((size_t)(b * 2048 + q0 + r)) * 2048 + cidx] = wa[i];
  }
}

// ---------------------------------------------------------------------------
// Kernel 4: output projection. out = raw(f16) @ Wout^T + bout  (fp32 out)
// grid = (8192/128, 1024/64), block = 256. Wout chunk staged in LDS.
// ---------------------------------------------------------------------------
__global__ void out_proj_kernel(const _Float16* __restrict__ raw,
                                const _Float16* __restrict__ Woh,  // (E,E) f16
                                const float* __restrict__ bout,    // (E)
                                float* __restrict__ out) {
  __shared__ __align__(16) _Float16 Ws[64 * 64];

  const int tid  = threadIdx.x;
  const int lane = tid & 31;
  const int wave = tid >> 5;
  const int m0 = blockIdx.x * 128 + wave * 16;
  const int n0 = blockIdx.y * 64;

  v8f acc[4] = {};

  const int arow = lane & 15;
  const int akb  = (lane & 16) ? 16 : 0;
  const int bn   = lane & 15;
  const int bkb  = (lane & 16) ? 16 : 0;

  for (int kc = 0; kc < 1024; kc += 64) {
    __syncthreads();
    for (int i = tid; i < 512; i += 256) {
      int r = i >> 3;
      int cc = (i & 7) * 8;
      *(uint4*)&Ws[r * 64 + cc] =
          *(const uint4*)(Woh + (size_t)(n0 + r) * 1024 + kc + cc);
    }
    __syncthreads();
#pragma unroll
    for (int kk = 0; kk < 64; kk += 32) {
      const _Float16* ap = raw + (size_t)(m0 + arow) * 1024 + kc + kk + akb;
      v16h a;
#pragma unroll
      for (int i = 0; i < 8; ++i) { a[i] = ap[i]; a[8 + i] = ap[8 + i]; }
      v16h bfr[4];
#pragma unroll
      for (int nt = 0; nt < 4; ++nt) {
        const _Float16* bp = Ws + (nt * 16 + bn) * 64 + kk + bkb;
#pragma unroll
        for (int i = 0; i < 16; ++i) bfr[nt][i] = bp[i];
      }
#pragma unroll
      for (int nt = 0; nt < 4; ++nt) acc[nt] = WMMA_F32_F16(a, bfr[nt], acc[nt]);
    }
  }

  const int crb = (lane & 16) ? 8 : 0;
#pragma unroll
  for (int nt = 0; nt < 4; ++nt) {
    int col = n0 + nt * 16 + (lane & 15);
    float bias = bout[col];
#pragma unroll
    for (int r = 0; r < 8; ++r) {
      int row = m0 + crb + r;
      out[(size_t)row * 1024 + col] = acc[nt][r] + bias;
    }
  }
}

// ---------------------------------------------------------------------------
extern "C" void kernel_launch(void* const* d_in, const int* in_sizes, int n_in,
                              void* d_out, int out_size, void* d_ws, size_t ws_size,
                              hipStream_t stream) {
  (void)in_sizes; (void)n_in; (void)out_size; (void)ws_size;

  const float* qin  = (const float*)d_in[0];
  const float* kin  = (const float*)d_in[1];
  const float* vin  = (const float*)d_in[2];
  const float* Win  = (const float*)d_in[3];   // (3E, E)
  const float* bin  = (const float*)d_in[4];   // (3E)
  const float* Wout = (const float*)d_in[5];   // (E, E)
  const float* bout = (const float*)d_in[6];   // (E)

  const size_t NBSE = (size_t)4 * 2048 * 1024;     // 8388608
  float* out  = (float*)d_out;                      // (B,S,E)
  float* wout = out + NBSE;                         // (B,Sq,Sk)

  // workspace carving (all f16)
  _Float16* Xhq = (_Float16*)d_ws;
  _Float16* Xhk = Xhq + NBSE;
  _Float16* Xhv = Xhk + NBSE;
  _Float16* Wh  = Xhv + NBSE;        // 3E*E   = 3145728
  _Float16* Woh = Wh + 3 * 1024 * 1024;
  _Float16* Qr  = Woh + 1024 * 1024; // (B,H,S,hd)
  _Float16* Kr  = Qr + NBSE;
  _Float16* Vr  = Kr + NBSE;
  _Float16* Vt  = Vr + NBSE;
  _Float16* rawb = Vt + NBSE;        // total ~143 MB

  // 0) convert everything to f16 once (all L2-resident afterwards)
  cvt_f16_kernel<<<(int)(NBSE / 256), 256, 0, stream>>>(qin, Xhq, (int)NBSE);
  cvt_f16_kernel<<<(int)(NBSE / 256), 256, 0, stream>>>(kin, Xhk, (int)NBSE);
  cvt_f16_kernel<<<(int)(NBSE / 256), 256, 0, stream>>>(vin, Xhv, (int)NBSE);
  cvt_f16_kernel<<<3 * 4096, 256, 0, stream>>>(Win, Wh, 3 * 1024 * 1024);
  cvt_f16_kernel<<<4096, 256, 0, stream>>>(Wout, Woh, 1024 * 1024);

  // 1) QKV projections (WMMA, LDS-staged weights), scatter to (B,H,S,hd) f16
  qkv_proj_kernel<<<dim3(64, 16, 3), 256, 0, stream>>>(
      Xhq, Xhk, Xhv, Wh, bin, Qr, Kr, Vr);

  // 2) RoPE in-place + V transpose
  rope_pack_kernel<<<16384, 256, 0, stream>>>(Qr, Kr, Vr, Vt);

  // 3) Fused attention with 267KB dynamic LDS (CDNA5: 320KB/WGP)
  const int SMEM_BYTES = 66848 * 4;
  (void)hipFuncSetAttribute((const void*)attention_kernel,
                            hipFuncAttributeMaxDynamicSharedMemorySize, SMEM_BYTES);
  attention_kernel<<<512, 256, SMEM_BYTES, stream>>>(Qr, Kr, Vt, rawb, wout);

  // 4) Output projection (WMMA, LDS-staged weights)
  out_proj_kernel<<<dim3(64, 16), 256, 0, stream>>>(rawb, Woh, bout, out);
}
